// UniCrossAtten_88373247082692
// MI455X (gfx1250) — matrix operand order repaired and checked by reading, not compiled, not run on hardware
//
#include <hip/hip_runtime.h>
#include <math.h>
#include <stdint.h>

typedef __attribute__((ext_vector_type(16))) __bf16 v16bf;
typedef __attribute__((ext_vector_type(8)))  float  v8f;
typedef __attribute__((ext_vector_type(4)))  int    v4i;

#define NQ 1800
#define BS 2
#define CC 256
#define NN 6
#define HH 180
#define WW 320

__device__ __forceinline__ float sigmoidf(float x) {
    return 1.0f / (1.0f + __expf(-x));
}

// ---------------------------------------------------------------------------
// Async global->LDS copy (16B) + async counter wait, with builtin/asm fallback
// (CDNA5 ISA: GLOBAL_LOAD_ASYNC_TO_LDS_B128, ASYNCcnt)
// ROCm clang-22 builtin takes typed AS pointers: (AS1 v4i*, AS3 v4i*, imm, imm)
// ---------------------------------------------------------------------------
__device__ __forceinline__ void async_copy_b128(const __bf16* g, __bf16* l) {
#if __has_builtin(__builtin_amdgcn_global_load_async_to_lds_b128)
    __builtin_amdgcn_global_load_async_to_lds_b128(
        (__attribute__((address_space(1))) v4i*)
            (__attribute__((address_space(1))) void*)(__bf16*)g,
        (__attribute__((address_space(3))) v4i*)
            (__attribute__((address_space(3))) void*)l,
        0, 0);
#else
    unsigned loff = (unsigned)(uintptr_t)(__attribute__((address_space(3))) char*)(char*)l;
    asm volatile("global_load_async_to_lds_b128 %0, %1, off"
                 :: "v"(loff), "v"(g) : "memory");
#endif
}

#if __has_builtin(__builtin_amdgcn_s_wait_asynccnt)
#define WAIT_ASYNC(n) __builtin_amdgcn_s_wait_asynccnt(n)
#else
#define WAIT_ASYNC(n) asm volatile("s_wait_asynccnt %0" :: "i"(n) : "memory")
#endif

// ---------------------------------------------------------------------------
// WMMA A-fragment packing (ISA 7.12.2, 16-bit A 16x32):
// lane = half*16 + mn holds, for K-tile kt, 16 contiguous bf16 at
//   Apacked[((mt*8 + kt)*32 + lane)*16 + e]
// where element e of row r=mt*16+mn, channel c=kt*32+cc32 satisfies
//   half = (cc32>>3)&1,  e = ((cc32>>4)<<3) | (cc32&7)
// ---------------------------------------------------------------------------
__device__ __forceinline__ size_t a_pack_idx(int r, int c) {
    int mt   = r >> 4;
    int mn   = r & 15;
    int kt   = c >> 5;
    int cc   = c & 31;
    int half = (cc >> 3) & 1;
    int e    = ((cc >> 4) << 3) | (cc & 7);
    return ((((size_t)mt * 8 + kt) * 32) + (half * 16 + mn)) * 16 + e;
}

// ---------------------------------------------------------------------------
// Pack f32 weight (K=256 x N=256, row-major) into bf16 WMMA B-fragment order:
//   Bpacked[((nt*8 + kt)*32 + lane)*16 + e] = W[k][n]
//   n = nt*16 + (lane&15),  k = kt*32 + (e>>3)*16 + (lane>>4)*8 + (e&7)
// ---------------------------------------------------------------------------
__global__ void pack_b_kernel(const float* __restrict__ src,
                              __bf16* __restrict__ dst) {
    int f    = blockIdx.x * 256 + threadIdx.x;   // 0 .. 65535
    int e    = f & 15;
    int lane = (f >> 4) & 31;
    int kt   = (f >> 9) & 7;
    int nt   = f >> 12;
    int n    = nt * 16 + (lane & 15);
    int k    = kt * 32 + (e >> 3) * 16 + (lane >> 4) * 8 + (e & 7);
    dst[f]   = (__bf16)src[(size_t)k * CC + n];
}

// ---------------------------------------------------------------------------
// Per-query: attention weights (sigmoid(q @ W_attn + b)) and pixel coords.
// One wave32 per (b,q); lanes stride channels, __shfl_xor reduction.
// ---------------------------------------------------------------------------
__global__ void attn_rp_kernel(const float* __restrict__ query,   // (nq,bs,C)
                               const float* __restrict__ refp,    // (bs,nq,3)
                               const float* __restrict__ W_attn,  // (C,N)
                               const float* __restrict__ b_attn,  // (N)
                               float* __restrict__ attn,          // (bs*nq, N)
                               float* __restrict__ px,
                               float* __restrict__ py) {
    int r = blockIdx.x;            // r = b*NQ + q
    int b = r / NQ;
    int q = r - b * NQ;
    int lane = threadIdx.x;        // 0..31 (wave32)

    float part[NN];
#pragma unroll
    for (int n = 0; n < NN; ++n) part[n] = 0.0f;

    const float* qrow = query + ((size_t)q * BS + b) * CC;
    for (int c = lane; c < CC; c += 32) {
        float qv = qrow[c];
        const float* wrow = W_attn + (size_t)c * NN;
#pragma unroll
        for (int n = 0; n < NN; ++n) part[n] += qv * wrow[n];
    }
#pragma unroll
    for (int off = 16; off > 0; off >>= 1) {
#pragma unroll
        for (int n = 0; n < NN; ++n) part[n] += __shfl_xor(part[n], off);
    }
    if (lane == 0) {
#pragma unroll
        for (int n = 0; n < NN; ++n)
            attn[(size_t)r * NN + n] = sigmoidf(part[n] + b_attn[n]);
        float gx = (sigmoidf(refp[(size_t)r * 3 + 0]) - 0.5f) * 2.0f;
        float gy = (sigmoidf(refp[(size_t)r * 3 + 1]) - 0.5f) * 2.0f;
        px[r] = ((gx + 1.0f) * (float)WW - 1.0f) * 0.5f;   // align_corners=False
        py[r] = ((gy + 1.0f) * (float)HH - 1.0f) * 0.5f;
    }
}

// ---------------------------------------------------------------------------
// Bilinear gather (zeros padding) + attention-weighted sum over N views.
// 256 threads = channels; result stored directly in WMMA A-fragment order.
// ---------------------------------------------------------------------------
__global__ void sample_kernel(const float* __restrict__ img,   // (bs,N,C,H,W)
                              const float* __restrict__ attn,  // (bs*nq, N)
                              const float* __restrict__ px,
                              const float* __restrict__ py,
                              __bf16* __restrict__ midp) {     // A-packed
    int r = blockIdx.x;            // b*NQ + q
    int b = r / NQ;
    int c = threadIdx.x;           // 0..255

    float x = px[r];
    float y = py[r];
    float x0f = floorf(x), y0f = floorf(y);
    int   x0 = (int)x0f, y0 = (int)y0f;
    float wx1 = x - x0f, wy1 = y - y0f;
    float w00 = (1.0f - wx1) * (1.0f - wy1);
    float w10 = wx1 * (1.0f - wy1);
    float w01 = (1.0f - wx1) * wy1;
    float w11 = wx1 * wy1;
    bool vx0 = (x0 >= 0) && (x0 < WW);
    bool vx1 = (x0 + 1 >= 0) && (x0 + 1 < WW);
    bool vy0 = (y0 >= 0) && (y0 < HH);
    bool vy1 = (y0 + 1 >= 0) && (y0 + 1 < HH);

    float acc = 0.0f;
#pragma unroll
    for (int n = 0; n < NN; ++n) {
        const float* base = img + (((size_t)(b * NN + n) * CC + c) * HH) * WW;
        float s = 0.0f;
        if (vy0 && vx0) s += w00 * base[y0 * WW + x0];
        if (vy0 && vx1) s += w10 * base[y0 * WW + x0 + 1];
        if (vy1 && vx0) s += w01 * base[(y0 + 1) * WW + x0];
        if (vy1 && vx1) s += w11 * base[(y0 + 1) * WW + x0 + 1];
        acc += attn[(size_t)r * NN + n] * s;
    }
    midp[a_pack_idx(r, c)] = (__bf16)acc;
}

// ---------------------------------------------------------------------------
// Pos-MLP layer 1: h1 = relu(LN(ref @ pe_w1 + pe_b1)); K=3 dot + block LN.
// Output bf16, WMMA A-fragment order.
// ---------------------------------------------------------------------------
__global__ void pe1_kernel(const float* __restrict__ refp,  // (bs,nq,3)
                           const float* __restrict__ w1,    // (3,C)
                           const float* __restrict__ b1,
                           const float* __restrict__ g1,
                           const float* __restrict__ bt1,
                           __bf16* __restrict__ h1p) {       // A-packed
    int r = blockIdx.x;
    int c = threadIdx.x;            // 0..255
    __shared__ float red[CC];

    float r0 = refp[(size_t)r * 3 + 0];
    float r1 = refp[(size_t)r * 3 + 1];
    float r2 = refp[(size_t)r * 3 + 2];
    float v = r0 * w1[c] + r1 * w1[CC + c] + r2 * w1[2 * CC + c] + b1[c];

    red[c] = v;
    __syncthreads();
#pragma unroll
    for (int s = CC / 2; s > 0; s >>= 1) {
        if (c < s) red[c] += red[c + s];
        __syncthreads();
    }
    float mean = red[0] * (1.0f / (float)CC);
    __syncthreads();

    float d = v - mean;
    red[c] = d * d;
    __syncthreads();
#pragma unroll
    for (int s = CC / 2; s > 0; s >>= 1) {
        if (c < s) red[c] += red[c + s];
        __syncthreads();
    }
    float var = red[0] * (1.0f / (float)CC);
    float o = d * rsqrtf(var + 1e-5f) * g1[c] + bt1[c];
    h1p[a_pack_idx(r, c)] = (__bf16)fmaxf(o, 0.0f);
}

// ---------------------------------------------------------------------------
// WMMA bf16 GEMM on fragment-packed operands, K = 256 (8 x k32 steps, fully
// unrolled). One wave32 per 16x16 tile. Tiles staged global->LDS with
// double-buffered GLOBAL_LOAD_ASYNC_TO_LDS_B128 (ASYNCcnt pipeline), consumed
// via ds_load_b128 into v_wmma_f32_16x16x32_bf16.
// ---------------------------------------------------------------------------
__global__ void wmma_gemm_packed(const __bf16* __restrict__ Ap,
                                 const __bf16* __restrict__ Bp,
                                 float* __restrict__ D) {     // (M,256) f32
    int lane = threadIdx.x & 31;
    int mt = blockIdx.y;            // 16-row tile
    int nt = blockIdx.x;            // 16-col tile

    __shared__ __align__(32) __bf16 lA[2][512];   // 1 KiB per buffer
    __shared__ __align__(32) __bf16 lB[2][512];

    const __bf16* aB = Ap + (size_t)mt * 8 * 512;   // 8 K-tiles x 512 frags
    const __bf16* bB = Bp + (size_t)nt * 8 * 512;

#define STAGE(kt, buf)                                                   \
    do {                                                                 \
        async_copy_b128(aB + (kt) * 512 + lane * 16,     &lA[buf][lane * 16]);     \
        async_copy_b128(aB + (kt) * 512 + lane * 16 + 8, &lA[buf][lane * 16 + 8]); \
        async_copy_b128(bB + (kt) * 512 + lane * 16,     &lB[buf][lane * 16]);     \
        async_copy_b128(bB + (kt) * 512 + lane * 16 + 8, &lB[buf][lane * 16 + 8]); \
    } while (0)

    STAGE(0, 0);
    v8f c = {};
#pragma unroll
    for (int kt = 0; kt < 8; ++kt) {
        int buf = kt & 1;
        if (kt < 7) {
            STAGE(kt + 1, buf ^ 1);     // overlap next tile's 4 async loads
            WAIT_ASYNC(4);              // current tile's loads complete
        } else {
            WAIT_ASYNC(0);
        }
        v16bf a = *(const v16bf*)&lA[buf][lane * 16];
        v16bf b = *(const v16bf*)&lB[buf][lane * 16];
        c = __builtin_amdgcn_wmma_f32_16x16x32_bf16(
                false, a, false, b, (short)0, c, false, false);
    }
#undef STAGE

    // D layout (ISA 7.12.2): VGPR r holds M = r + half*8, N = lane%16
    int mn = lane & 15, half = lane >> 4;
#pragma unroll
    for (int rI = 0; rI < 8; ++rI)
        D[(size_t)(mt * 16 + rI + half * 8) * CC + nt * 16 + mn] = c[rI];
}

// ---------------------------------------------------------------------------
// Pos-MLP layer 2 epilogue: h = relu(LN(h2 + pe_b2)), in place (f32).
// ---------------------------------------------------------------------------
__global__ void ln2_kernel(float* __restrict__ h2,     // (bs*nq, C) in/out
                           const float* __restrict__ b2,
                           const float* __restrict__ g2,
                           const float* __restrict__ bt2) {
    int r = blockIdx.x;
    int c = threadIdx.x;
    __shared__ float red[CC];

    float v = h2[(size_t)r * CC + c] + b2[c];

    red[c] = v;
    __syncthreads();
#pragma unroll
    for (int s = CC / 2; s > 0; s >>= 1) {
        if (c < s) red[c] += red[c + s];
        __syncthreads();
    }
    float mean = red[0] * (1.0f / (float)CC);
    __syncthreads();

    float d = v - mean;
    red[c] = d * d;
    __syncthreads();
#pragma unroll
    for (int s = CC / 2; s > 0; s >>= 1) {
        if (c < s) red[c] += red[c + s];
        __syncthreads();
    }
    float var = red[0] * (1.0f / (float)CC);
    float o = d * rsqrtf(var + 1e-5f) * g2[c] + bt2[c];
    h2[(size_t)r * CC + c] = fmaxf(o, 0.0f);
}

// ---------------------------------------------------------------------------
// Final fused add with (bs,nq,C) -> (nq,bs,C) transpose.
// ---------------------------------------------------------------------------
__global__ void final_kernel(const float* __restrict__ out2,    // (bs*nq, C)
                             const float* __restrict__ bout,    // (C)
                             const float* __restrict__ query,   // (nq,bs,C)
                             const float* __restrict__ posfeat, // (bs*nq, C)
                             float* __restrict__ out) {         // (nq,bs,C)
    int r = blockIdx.x;            // b*NQ + q
    int b = r / NQ;
    int q = r - b * NQ;
    int c = threadIdx.x;
    size_t oi = ((size_t)q * BS + b) * CC + c;
    out[oi] = out2[(size_t)r * CC + c] + bout[c] + query[oi]
            + posfeat[(size_t)r * CC + c];
}

// ---------------------------------------------------------------------------
extern "C" void kernel_launch(void* const* d_in, const int* in_sizes, int n_in,
                              void* d_out, int out_size, void* d_ws, size_t ws_size,
                              hipStream_t stream) {
    const float* query  = (const float*)d_in[0];
    const float* refp   = (const float*)d_in[1];
    const float* img    = (const float*)d_in[2];
    const float* W_attn = (const float*)d_in[3];
    const float* b_attn = (const float*)d_in[4];
    const float* W_out  = (const float*)d_in[5];
    const float* b_out  = (const float*)d_in[6];
    const float* pe_w1  = (const float*)d_in[7];
    const float* pe_b1  = (const float*)d_in[8];
    const float* pe_g1  = (const float*)d_in[9];
    const float* pe_bt1 = (const float*)d_in[10];
    const float* pe_w2  = (const float*)d_in[11];
    const float* pe_b2  = (const float*)d_in[12];
    const float* pe_g2  = (const float*)d_in[13];
    const float* pe_bt2 = (const float*)d_in[14];
    float* out = (float*)d_out;

    const int R = BS * NQ;          // 3600 rows (= 225 WMMA row-tiles)

    // workspace carve-up (256B aligned)
    char* base = (char*)d_ws;
    size_t off = 0;
    auto carve = [&](size_t bytes) -> char* {
        char* p = base + off;
        off = (off + bytes + 255) & ~(size_t)255;
        return p;
    };
    float*  ws_attn = (float*) carve((size_t)R * NN * sizeof(float));
    float*  ws_px   = (float*) carve((size_t)R * sizeof(float));
    float*  ws_py   = (float*) carve((size_t)R * sizeof(float));
    __bf16* ws_midp = (__bf16*)carve((size_t)R * CC * sizeof(__bf16)); // A-packed
    __bf16* ws_h1p  = (__bf16*)carve((size_t)R * CC * sizeof(__bf16)); // A-packed
    float*  ws_out2 = (float*) carve((size_t)R * CC * sizeof(float));
    float*  ws_h2   = (float*) carve((size_t)R * CC * sizeof(float));
    __bf16* ws_woB  = (__bf16*)carve((size_t)CC * CC * sizeof(__bf16)); // B-packed
    __bf16* ws_w2B  = (__bf16*)carve((size_t)CC * CC * sizeof(__bf16)); // B-packed
    (void)ws_size;

    // 1. weight conversion + fragment packing for WMMA
    pack_b_kernel<<<CC, 256, 0, stream>>>(W_out, ws_woB);
    pack_b_kernel<<<CC, 256, 0, stream>>>(pe_w2, ws_w2B);

    // 2. attention weights + pixel coords (one wave32 per query)
    attn_rp_kernel<<<R, 32, 0, stream>>>(query, refp, W_attn, b_attn,
                                         ws_attn, ws_px, ws_py);

    // 3. bilinear gather + weighted view sum  (dominant HBM traffic)
    sample_kernel<<<R, CC, 0, stream>>>(img, ws_attn, ws_px, ws_py, ws_midp);

    // 4. out2 = mid @ W_out   (async-LDS staged bf16 WMMA, f32 accumulate)
    {
        dim3 grid(CC / 16, R / 16);
        wmma_gemm_packed<<<grid, 32, 0, stream>>>(ws_midp, ws_woB, ws_out2);
    }

    // 5. pos-MLP layer 1
    pe1_kernel<<<R, CC, 0, stream>>>(refp, pe_w1, pe_b1, pe_g1, pe_bt1, ws_h1p);

    // 6. h2 = h1 @ pe_w2      (async-LDS staged bf16 WMMA, f32 accumulate)
    {
        dim3 grid(CC / 16, R / 16);
        wmma_gemm_packed<<<grid, 32, 0, stream>>>(ws_h1p, ws_w2B, ws_h2);
    }

    // 7. pos-MLP layer 2 epilogue (LN + relu, in place)
    ln2_kernel<<<R, CC, 0, stream>>>(ws_h2, pe_b2, pe_g2, pe_bt2);

    // 8. final fused add + transpose to (nq,bs,C)
    final_kernel<<<R, CC, 0, stream>>>(ws_out2, b_out, query, ws_h2, out);
    (void)out_size; (void)n_in; (void)in_sizes;
}
